// DLRMs_7705171329793
// MI455X (gfx1250) — matrix operand order repaired and checked by reading, not compiled, not run on hardware
//
#include <hip/hip_runtime.h>

// ---------------- problem constants (match reference) ----------------
#define B_TOTAL       16384
#define NUM_SPARSE    26
#define EMB           64
#define SPARSE_VOCAB  100000
#define HIDDEN        128
#define NSEG          (NUM_SPARSE + 2)   // 28 embedding segments of K=64
#define MLP_IN        (NSEG * EMB)       // 1792
#define ROWS_PER_WG   64
#define THREADS       128                // 4 wave32
#define APITCH        72                 // halfs per A row: 144B (16B aligned, bank-spread)
#define WPITCH        72                 // halfs per W col
#define WSEG_HALFS    (EMB * HIDDEN)     // 8192 f16 per segment in workspace

typedef __attribute__((ext_vector_type(16))) _Float16 v16h;
typedef __attribute__((ext_vector_type(8)))  float    v8f;

union Frag32 { uint4 q[2]; v16h v; };

__device__ __forceinline__ unsigned pack2f16(float a, float b) {
  union { _Float16 h; unsigned short u; } ca, cb;
  ca.h = (_Float16)a;
  cb.h = (_Float16)b;
  return (unsigned)ca.u | ((unsigned)cb.u << 16);
}

// CDNA5 async global->LDS copy (per-lane LDS dest, tracked by ASYNCcnt)
__device__ __forceinline__ void async_g2l_b128(unsigned lds_byte_addr, const void* gaddr) {
  asm volatile("global_load_async_to_lds_b128 %0, %1, off"
               :: "v"(lds_byte_addr), "v"((unsigned long long)gaddr)
               : "memory");
}
__device__ __forceinline__ void wait_asynccnt0() {
  asm volatile("s_wait_asynccnt 0x0" ::: "memory");
}

// ---------- prologue: W1 fp32 [K=1792][N=128] -> f16 workspace [seg][col][k] ----------
__global__ __launch_bounds__(256)
void prep_w1(const float* __restrict__ W1, _Float16* __restrict__ wsW) {
  const int u = blockIdx.x * blockDim.x + threadIdx.x;     // one k-pair per thread
  if (u >= (MLP_IN * HIDDEN) / 2) return;
  const int kp  = u & 31;                 // k = 2*kp within a column
  const int col = (u >> 5) & (HIDDEN - 1);
  const int seg = u >> 12;                // / (32*128)
  const int k   = kp * 2;
  const float w0 = W1[(size_t)(seg * EMB + k)     * HIDDEN + col];
  const float w1 = W1[(size_t)(seg * EMB + k + 1) * HIDDEN + col];
  unsigned* dst = reinterpret_cast<unsigned*>(
      wsW + (size_t)seg * WSEG_HALFS + (size_t)col * EMB + k);
  *dst = pack2f16(w0, w1);
}

// ---------- helpers for the fused kernel ----------
__device__ __forceinline__ void stageA(_Float16* abuf, const float* src, int r, int part) {
  const float4* s4 = reinterpret_cast<const float4*>(src) + part * 8;
  _Float16* adst = abuf + (size_t)r * APITCH + part * 32;
  #pragma unroll
  for (int j = 0; j < 8; ++j) {
    const float4 v = s4[j];
    uint2 p;
    p.x = pack2f16(v.x, v.y);
    p.y = pack2f16(v.z, v.w);
    *reinterpret_cast<uint2*>(adst + j * 4) = p;
  }
}

__device__ __forceinline__ void stageW_async(_Float16* wbuf, const _Float16* wseg, int col) {
  const unsigned lbase = (unsigned)(size_t)(void*)(wbuf + (size_t)col * WPITCH);
  const char* g = reinterpret_cast<const char*>(wseg + (size_t)col * EMB);
  #pragma unroll
  for (int j = 0; j < 8; ++j)
    async_g2l_b128(lbase + j * 16, g + j * 16);
}

__device__ __forceinline__ void loadB(Frag32& b, const _Float16* wbuf,
                                      int s, int n, int lane15, int halfsel) {
  const _Float16* p = wbuf + (size_t)((n << 4) + lane15) * WPITCH + s * 32 + halfsel * 8;
  b.q[0] = *reinterpret_cast<const uint4*>(p);
  b.q[1] = *reinterpret_cast<const uint4*>(p + 16);
}

__global__ __launch_bounds__(THREADS)
void dlrm_fused_wmma(const int*   __restrict__ user_ids,
                     const int*   __restrict__ item_ids,
                     const int*   __restrict__ sparse_idx,
                     const float* __restrict__ user_emb,
                     const float* __restrict__ item_emb,
                     const float* __restrict__ sparse_tables,
                     const _Float16* __restrict__ wsW,   // pre-converted W1
                     const float* __restrict__ b1,
                     const float* __restrict__ W2,
                     const float* __restrict__ b2,
                     float*       __restrict__ out)
{
  __shared__ _Float16 lsA[2][ROWS_PER_WG * APITCH];  // 2 x  9216 B
  __shared__ _Float16 lsW[2][HIDDEN      * WPITCH];  // 2 x 18432 B

  const int t       = threadIdx.x;
  const int lane    = t & 31;
  const int lane15  = lane & 15;
  const int wave    = t >> 5;                 // 0..3, each owns 16 rows
  const int rowbase = blockIdx.x * ROWS_PER_WG;

  const int gr      = rowbase + (t >> 1);     // gather row (2 threads per row)
  const int part    = t & 1;

  const int halfsel = (lane >= 16) ? 1 : 0;
  const int arow    = (wave << 4) + lane15;

  auto row_src = [&](int seg) -> const float* {
    if (seg == 0) return user_emb + (size_t)user_ids[gr] * EMB;
    if (seg == 1) return item_emb + (size_t)item_ids[gr] * EMB;
    const int f = seg - 2;
    const size_t idx = (size_t)sparse_idx[(size_t)gr * NUM_SPARSE + f];
    return sparse_tables + ((size_t)f * SPARSE_VOCAB + idx) * EMB;
  };

  v8f c[8] = {};

  // ---- prologue of the pipeline: stage segment 0, prefetch segment 1 ----
  stageW_async(lsW[0], wsW, t);
  stageA(lsA[0], row_src(0), t >> 1, part);
  const float* nsrc = row_src(1);
  __builtin_prefetch(nsrc + part * 32, 0, 0);     // global_prefetch_b8

  for (int seg = 0; seg < NSEG; ++seg) {
    wait_asynccnt0();          // this wave's async W copies for buffer (seg&1)
    __syncthreads();           // everyone's staging of buffer (seg&1) visible
    const int cb = seg & 1;

    // ---- stage NEXT segment into the other buffer (overlaps with WMMA) ----
    if (seg + 1 < NSEG) {
      stageW_async(lsW[cb ^ 1], wsW + (size_t)(seg + 1) * WSEG_HALFS, t);
      stageA(lsA[cb ^ 1], nsrc, t >> 1, part);
      if (seg + 2 < NSEG) {
        nsrc = row_src(seg + 2);
        __builtin_prefetch(nsrc + part * 32, 0, 0);
      }
    }

    // ---- compute this wave's 16 rows x 128 cols, K += 64 ----
    Frag32 a0, a1;
    {
      const _Float16* ab = lsA[cb] + (size_t)arow * APITCH + halfsel * 8;
      a0.q[0] = *reinterpret_cast<const uint4*>(ab);
      a0.q[1] = *reinterpret_cast<const uint4*>(ab + 16);
      a1.q[0] = *reinterpret_cast<const uint4*>(ab + 32);
      a1.q[1] = *reinterpret_cast<const uint4*>(ab + 48);
    }
    Frag32 bf[2];
    loadB(bf[0], lsW[cb], 0, 0, lane15, halfsel);
    #pragma unroll
    for (int i = 0; i < 16; ++i) {         // i = s*8 + n, B frags ping-pong
      const int s = i >> 3, n = i & 7;
      if (i < 15)
        loadB(bf[(i + 1) & 1], lsW[cb], (i + 1) >> 3, (i + 1) & 7, lane15, halfsel);
      const Frag32& af = (s == 0) ? a0 : a1;
      c[n] = __builtin_amdgcn_wmma_f32_16x16x32_f16(
          false, af.v, false, bf[i & 1].v, (short)0, c[n], false, false);
    }
  }

  // ---- layer 2 fused in-register: h = relu(c + b1); out = sigmoid(h . W2 + b2) ----
  float acc[8];
  #pragma unroll
  for (int v = 0; v < 8; ++v) acc[v] = 0.0f;
  #pragma unroll
  for (int n = 0; n < 8; ++n) {
    const int col = (n << 4) + lane15;
    const float bb = b1[col];
    const float w2 = W2[col];
    #pragma unroll
    for (int v = 0; v < 8; ++v) {
      float h = c[n][v] + bb;
      h = fmaxf(h, 0.0f);
      acc[v] += h * w2;
    }
  }
  #pragma unroll
  for (int off = 1; off < 16; off <<= 1) {
    #pragma unroll
    for (int v = 0; v < 8; ++v)
      acc[v] += __shfl_xor(acc[v], off, 32);
  }
  if (lane15 == 0) {
    const int   grow  = rowbase + (wave << 4) + halfsel * 8;
    const float bias2 = b2[0];
    #pragma unroll
    for (int v = 0; v < 8; ++v) {
      const float x = acc[v] + bias2;
      out[grow + v] = 1.0f / (1.0f + __expf(-x));
    }
  }
}

extern "C" void kernel_launch(void* const* d_in, const int* in_sizes, int n_in,
                              void* d_out, int out_size, void* d_ws, size_t ws_size,
                              hipStream_t stream) {
  (void)in_sizes; (void)n_in; (void)out_size; (void)ws_size;
  const int*   user_ids      = (const int*)  d_in[0];
  const int*   item_ids      = (const int*)  d_in[1];
  const int*   sparse_feat   = (const int*)  d_in[2];
  const float* user_emb      = (const float*)d_in[3];
  const float* item_emb      = (const float*)d_in[4];
  const float* sparse_tables = (const float*)d_in[5];
  const float* W1            = (const float*)d_in[6];
  const float* b1            = (const float*)d_in[7];
  const float* W2            = (const float*)d_in[8];
  const float* b2            = (const float*)d_in[9];
  float*       out           = (float*)d_out;
  _Float16*    wsW           = (_Float16*)d_ws;    // 448 KB: f16 W1, [seg][col][k]

  // 1) one-time (per launch) W1 convert+transpose into workspace
  {
    const int pairs = (MLP_IN * HIDDEN) / 2;       // 114688
    hipLaunchKernelGGL(prep_w1, dim3((pairs + 255) / 256), dim3(256), 0, stream,
                       W1, wsW);
  }
  // 2) fused gather + GEMM1(WMMA) + ReLU + GEMM2 + sigmoid
  hipLaunchKernelGGL(dlrm_fused_wmma, dim3(B_TOTAL / ROWS_PER_WG), dim3(THREADS),
                     0, stream,
                     user_ids, item_ids, sparse_feat,
                     user_emb, item_emb, sparse_tables,
                     wsW, b1, W2, b2, out);
}
